// treelstm_27986006900855
// MI455X (gfx1250) — matrix operand order repaired and checked by reading, not compiled, not run on hardware
//
#include <hip/hip_runtime.h>

// TreeLSTM cell, fully fused, bf16 WMMA on gfx1250 (wave32), with async
// global->LDS double-buffered weight staging.
// d_ws: 8 weight matrices as bf16 [gate][out=512][in=512], gate order mirrors
// d_in order: 0=W_ix 1=W_fx 2=W_ux 3=W_ox 4=W_ih 5=W_fh 6=W_uh 7=W_oh (4 MiB).

typedef __attribute__((ext_vector_type(16))) __bf16 v16bf;
typedef __attribute__((ext_vector_type(8)))  float  v8f;
typedef int vi4 __attribute__((vector_size(16)));   // 16B chunk type for async builtin

#define DIMK 512
#define NB   32768
#define NCH  4
#define WELEMS (DIMK * DIMK)

// LDS staging geometry: per stage, 8 gates x 32 cols x 128 K of bf16.
// Row = one (gate,col) pair: 128 bf16 = 256B payload, padded to 272B (68 dwords)
// so the 16 column-lanes land on distinct bank groups (stride 4 banks).
// 32B chunks within a row are XOR-swizzled by (col&7) -> conflict-free b128 reads.
#define KCH      128
#define ROWB     272
#define NROWS    256                   // 8 gates * 32 cols
#define BUFBYTES (NROWS * ROWB)        // 69632
#define LDSBYTES (2 * BUFBYTES)        // 139264

#if __has_builtin(__builtin_amdgcn_global_load_async_to_lds_b128)
#define HAVE_ASYNC_LDS 1
#else
#define HAVE_ASYNC_LDS 0
#endif

struct WPtrs { const float* p[8]; };
struct BPtrs { const float* b[8]; };

// ---------------- weight f32 -> bf16 conversion ------------------------------------
__global__ void convert_weights_kernel(WPtrs wp, __bf16* __restrict__ out) {
    unsigned int idx = blockIdx.x * blockDim.x + threadIdx.x;
    unsigned int g   = idx >> 18;
    unsigned int off = idx & (WELEMS - 1);
    out[idx] = (__bf16)wp.p[g][off];
}

// ---------------- helpers ----------------------------------------------------------
__device__ __forceinline__ float sigmoid_f(float x) {
    return 1.0f / (1.0f + __expf(-x));
}
__device__ __forceinline__ float tanh_f(float x) {
    float t = __expf(-2.0f * __builtin_fabsf(x));
    float r = (1.0f - t) / (1.0f + t);
    return __builtin_copysignf(r, x);
}

template <int N> __device__ __forceinline__ void wait_async() {
#if HAVE_ASYNC_LDS
#if __has_builtin(__builtin_amdgcn_s_wait_asynccnt)
    __builtin_amdgcn_s_wait_asynccnt(N);
#else
    asm volatile("s_wait_asynccnt %0" ::"i"(N));
#endif
#endif
}

__device__ __forceinline__ void async_copy16(const __bf16* src, char* dst) {
#if HAVE_ASYNC_LDS
    vi4* g = reinterpret_cast<vi4*>(const_cast<__bf16*>(src));
    vi4* l = reinterpret_cast<vi4*>(dst);
    __builtin_amdgcn_global_load_async_to_lds_b128(g, l, 0, 0);
#else
    *reinterpret_cast<uint4*>(dst) = *reinterpret_cast<const uint4*>(src);
#endif
}

// Stage one (coltile, kchunk) weight slab into LDS buffer `buf`.
// 4096 16B-chunks; each of 256 threads moves 16 chunks.
__device__ __forceinline__ void stage_weights(const __bf16* __restrict__ wbf,
                                              char* __restrict__ smem,
                                              int buf, int colbase, int kc, int tid) {
    char* base = smem + buf * BUFBYTES;
#pragma unroll
    for (int i = 0; i < 16; ++i) {
        int chunk = tid + i * 256;
        int j     = chunk & 15;        // 16B chunk within row (8 bf16)
        int row   = chunk >> 4;        // g*32 + c32
        int c32   = row & 31;
        int g     = row >> 5;
        int pp    = (j >> 1) ^ (c32 & 7);                 // swizzled 32B-pair index
        char* dst = base + row * ROWB + (pp * 2 + (j & 1)) * 16;
        const __bf16* src = wbf + (size_t)g * WELEMS +
                            (size_t)(colbase + c32) * DIMK + kc * KCH + j * 8;
        async_copy16(src, dst);
    }
}

// B-fragment (32x16 bf16, K x N) from LDS: lane = column (lane&15), half = lane>>4.
__device__ __forceinline__ v16bf load_b_lds(const char* __restrict__ smem, int buf,
                                            int g, int c32, int kbrel, int half) {
    int pp = (kbrel * 2 + half) ^ (c32 & 7);
    const uint4* p = reinterpret_cast<const uint4*>(
        smem + buf * BUFBYTES + (g * 32 + c32) * ROWB + pp * 32);
    union { uint4 q[2]; v16bf v; } u;
    u.q[0] = p[0];
    u.q[1] = p[1];
    return u.v;
}

// A-fragment (16x32 bf16): load f32 rows from global, convert to bf16 in regs.
__device__ __forceinline__ v16bf load_a_frag(const float* __restrict__ rowptr,
                                             int kbase, int half) {
    const float4* p0 = reinterpret_cast<const float4*>(rowptr + kbase + half * 16);
    const float4* p1 = reinterpret_cast<const float4*>(rowptr + kbase + half * 16 + 8);
    float4 q0 = p0[0], q1 = p0[1];
    float4 q2 = p1[0], q3 = p1[1];
    v16bf v;
    v[0]  = (__bf16)q0.x; v[1]  = (__bf16)q0.y; v[2]  = (__bf16)q0.z; v[3]  = (__bf16)q0.w;
    v[4]  = (__bf16)q1.x; v[5]  = (__bf16)q1.y; v[6]  = (__bf16)q1.z; v[7]  = (__bf16)q1.w;
    v[8]  = (__bf16)q2.x; v[9]  = (__bf16)q2.y; v[10] = (__bf16)q2.z; v[11] = (__bf16)q2.w;
    v[12] = (__bf16)q3.x; v[13] = (__bf16)q3.y; v[14] = (__bf16)q3.z; v[15] = (__bf16)q3.w;
    return v;
}

__device__ __forceinline__ v8f wmma_bf16(v16bf a, v16bf b, v8f c) {
    return __builtin_amdgcn_wmma_f32_16x16x32_bf16(false, a, false, b,
                                                   (short)0, c, false, false);
}

// ---------------- fused TreeLSTM kernel --------------------------------------------
// grid: (NB/128, 2), block: 256 threads (8 waves).
// Block: 128 rows x 256 cols. Wave w owns rows [w*16, w*16+16).
// 32 pipeline stages = 8 coltiles x 4 K-chunks, double-buffered in LDS.
__launch_bounds__(256, 1)
__global__ void treelstm_wmma_kernel(const float* __restrict__ inputs,
                                     const float* __restrict__ child_c,
                                     const float* __restrict__ child_h,
                                     const float* __restrict__ hsum,
                                     const __bf16* __restrict__ wbf,
                                     BPtrs bp,
                                     float* __restrict__ out_c,
                                     float* __restrict__ out_h) {
    extern __shared__ char smem[];

    const int tid     = threadIdx.x;
    const int lane    = tid & 31;
    const int wave    = tid >> 5;
    const int lrow    = lane & 15;
    const int half    = lane >> 4;
    const int rowbase = blockIdx.x * 128 + wave * 16;
    const int colgrp  = blockIdx.y * 256;

    const int arow = rowbase + lrow;
    const float* xrow = inputs + (size_t)arow * DIMK;
    const float* hrow = hsum   + (size_t)arow * DIMK;
    const float* crow[NCH];
#pragma unroll
    for (int k = 0; k < NCH; ++k)
        crow[k] = child_h + ((size_t)k * NB + arow) * DIMK;

    // prologue: start stage 0
    stage_weights(wbf, smem, 0, colgrp, 0, tid);

    v8f acc[2][8];   // [col subtile][gate]: 0=i 1=u 2=o 3=fx 4..7=fh[k]

    for (int st = 0; st < 32; ++st) {
        const int ct      = st >> 2;
        const int kc      = st & 3;
        const int buf     = st & 1;
        const int colbase = colgrp + ct * 32;

        if (kc == 0) {
#pragma unroll
            for (int s = 0; s < 2; ++s)
#pragma unroll
                for (int g = 0; g < 8; ++g)
                    acc[s][g] = (v8f){0.f, 0.f, 0.f, 0.f, 0.f, 0.f, 0.f, 0.f};
        }

        // issue async copy for next stage into the other buffer
        if (st + 1 < 32) {
            stage_weights(wbf, smem, (st + 1) & 1, colgrp + ((st + 1) >> 2) * 32,
                          (st + 1) & 3, tid);
            wait_async<16>();   // this stage's 16 async ops (issued last iter) done
        } else {
            wait_async<0>();
        }
        __syncthreads();        // all waves' copies for this stage visible

        for (int kbrel = 0; kbrel < 4; ++kbrel) {
            const int kbase = kc * KCH + kbrel * 32;

            __builtin_prefetch(xrow + kbase + 32, 0, 3);
            __builtin_prefetch(hrow + kbase + 32, 0, 3);
#pragma unroll
            for (int k = 0; k < NCH; ++k)
                __builtin_prefetch(crow[k] + kbase + 32, 0, 3);

            v16bf Ax = load_a_frag(xrow, kbase, half);
            v16bf Ah = load_a_frag(hrow, kbase, half);
            v16bf Ac[NCH];
#pragma unroll
            for (int k = 0; k < NCH; ++k)
                Ac[k] = load_a_frag(crow[k], kbase, half);

#pragma unroll
            for (int s = 0; s < 2; ++s) {
                const int c32 = s * 16 + lrow;
                acc[s][0] = wmma_bf16(Ax, load_b_lds(smem, buf, 0, c32, kbrel, half), acc[s][0]);
                acc[s][0] = wmma_bf16(Ah, load_b_lds(smem, buf, 4, c32, kbrel, half), acc[s][0]);
                acc[s][1] = wmma_bf16(Ax, load_b_lds(smem, buf, 2, c32, kbrel, half), acc[s][1]);
                acc[s][1] = wmma_bf16(Ah, load_b_lds(smem, buf, 6, c32, kbrel, half), acc[s][1]);
                acc[s][2] = wmma_bf16(Ax, load_b_lds(smem, buf, 3, c32, kbrel, half), acc[s][2]);
                acc[s][2] = wmma_bf16(Ah, load_b_lds(smem, buf, 7, c32, kbrel, half), acc[s][2]);
                acc[s][3] = wmma_bf16(Ax, load_b_lds(smem, buf, 1, c32, kbrel, half), acc[s][3]);
                v16bf Bfh = load_b_lds(smem, buf, 5, c32, kbrel, half);
#pragma unroll
                for (int k = 0; k < NCH; ++k)
                    acc[s][4 + k] = wmma_bf16(Ac[k], Bfh, acc[s][4 + k]);
            }
        }
        __syncthreads();        // everyone done reading buf before it is restaged

        if (kc == 3) {
            // ---------------- epilogue for this 32-col tile ------------------------
#pragma unroll
            for (int s = 0; s < 2; ++s) {
                const int col = colbase + s * 16 + lrow;
                const float bi = bp.b[0][col] + bp.b[4][col];
                const float bf = bp.b[1][col] + bp.b[5][col];
                const float bu = bp.b[2][col] + bp.b[6][col];
                const float bo = bp.b[3][col] + bp.b[7][col];
#pragma unroll
                for (int r = 0; r < 8; ++r) {
                    const int row = rowbase + r + half * 8;
                    float iv = sigmoid_f(acc[s][0][r] + bi);
                    float uv = tanh_f   (acc[s][1][r] + bu);
                    float ov = sigmoid_f(acc[s][2][r] + bo);
                    float fx = acc[s][3][r] + bf;
                    float cs = iv * uv;
#pragma unroll
                    for (int k = 0; k < NCH; ++k) {
                        float fk = sigmoid_f(acc[s][4 + k][r] + fx);
                        cs += fk * child_c[((size_t)k * NB + row) * DIMK + col];
                    }
                    out_c[(size_t)row * DIMK + col] = cs;
                    out_h[(size_t)row * DIMK + col] = ov * tanh_f(cs);
                }
            }
        }
    }
}

// ---------------- host launcher ----------------------------------------------------
extern "C" void kernel_launch(void* const* d_in, const int* in_sizes, int n_in,
                              void* d_out, int out_size, void* d_ws, size_t ws_size,
                              hipStream_t stream) {
    (void)in_sizes; (void)n_in; (void)out_size; (void)ws_size;

    const float* inputs  = (const float*)d_in[0];
    const float* child_c = (const float*)d_in[1];
    const float* child_h = (const float*)d_in[2];
    const float* hsum    = (const float*)d_in[3];

    WPtrs wp;
    wp.p[0] = (const float*)d_in[4];   // W_ix
    wp.p[1] = (const float*)d_in[6];   // W_fx
    wp.p[2] = (const float*)d_in[8];   // W_ux
    wp.p[3] = (const float*)d_in[10];  // W_ox
    wp.p[4] = (const float*)d_in[12];  // W_ih
    wp.p[5] = (const float*)d_in[14];  // W_fh
    wp.p[6] = (const float*)d_in[16];  // W_uh
    wp.p[7] = (const float*)d_in[18];  // W_oh

    BPtrs bp;
    bp.b[0] = (const float*)d_in[5];   // b_ix
    bp.b[1] = (const float*)d_in[7];   // b_fx
    bp.b[2] = (const float*)d_in[9];   // b_ux
    bp.b[3] = (const float*)d_in[11];  // b_ox
    bp.b[4] = (const float*)d_in[13];  // b_ih
    bp.b[5] = (const float*)d_in[15];  // b_fh
    bp.b[6] = (const float*)d_in[17];  // b_uh
    bp.b[7] = (const float*)d_in[19];  // b_oh

    __bf16* wbf = (__bf16*)d_ws;       // 4 MiB bf16 weights

    convert_weights_kernel<<<(8 * WELEMS) / 256, 256, 0, stream>>>(wp, wbf);

    // allow the large dynamic-LDS allocation (136 KB of 320 KB per WGP)
    static bool attr_set = false;
    if (!attr_set) {
        (void)hipFuncSetAttribute((const void*)treelstm_wmma_kernel,
                                  hipFuncAttributeMaxDynamicSharedMemorySize, LDSBYTES);
        attr_set = true;
    }

    float* out_c = (float*)d_out;
    float* out_h = out_c + (size_t)NB * DIMK;
    dim3 grid(NB / 128, 2);
    treelstm_wmma_kernel<<<grid, 256, LDSBYTES, stream>>>(inputs, child_c, child_h,
                                                          hsum, wbf, bp, out_c, out_h);
}
